// MultiHeadAttention_11441792877228
// MI455X (gfx1250) — matrix operand order repaired and checked by reading, not compiled, not run on hardware
//
#include <hip/hip_runtime.h>
#include <hip/hip_bf16.h>
#include <math.h>

// Problem constants (match reference)
#define E_DIM 1024
#define NH    16
#define DH    64
#define BATCH 4
#define SEQ   2048
#define MTOT  (BATCH * SEQ)   // 8192 rows

typedef __attribute__((ext_vector_type(16))) __bf16 v16bf;
typedef __attribute__((ext_vector_type(8)))  __bf16 v8bf;
typedef __attribute__((ext_vector_type(8)))  float  v8f;

static __device__ __forceinline__ v8f wmma_bf16(v16bf a, v16bf b, v8f c) {
  // D = A(16x32) * B(32x16) + C, f32 accumulate
  return __builtin_amdgcn_wmma_f32_16x16x32_bf16(false, a, false, b, (short)0, c,
                                                 false, false);
}

static __device__ __forceinline__ v16bf cat8(v8bf lo, v8bf hi) {
  return __builtin_shufflevector(lo, hi, 0, 1, 2, 3, 4, 5, 6, 7,
                                         8, 9, 10, 11, 12, 13, 14, 15);
}

// ---------------------------------------------------------------------------
// Async global->LDS copy (CDNA5 GLOBAL_LOAD_ASYNC_TO_LDS_B128, ASYNCcnt).
// Probe-derived signature: (v4i addrspace(1)*, v4i addrspace(3)*, imm, imm).
// Guarded so we fall back to the VGPR-staged path if the builtin is absent.
// ---------------------------------------------------------------------------
#if defined(__has_builtin)
#if __has_builtin(__builtin_amdgcn_global_load_async_to_lds_b128)
#define HAVE_ASYNC_LDS 1
#endif
#endif

#ifdef HAVE_ASYNC_LDS
typedef int v4i __attribute__((vector_size(4 * sizeof(int))));
typedef __attribute__((address_space(1))) v4i* g_v4i_ptr;
typedef __attribute__((address_space(3))) v4i* l_v4i_ptr;

static __device__ __forceinline__ void async_b128(void* lds, const void* g) {
  __builtin_amdgcn_global_load_async_to_lds_b128((g_v4i_ptr)g, (l_v4i_ptr)lds, 0, 0);
}
static __device__ __forceinline__ void async_wait0() {
#if __has_builtin(__builtin_amdgcn_s_wait_asynccnt)
  __builtin_amdgcn_s_wait_asynccnt(0);
#else
  asm volatile("s_wait_asynccnt 0" ::: "memory");
#endif
}
#endif

static __device__ __forceinline__ void copy16_bf16(__bf16* lds, const __bf16* g) {
#ifdef HAVE_ASYNC_LDS
  async_b128(lds, g);
  async_b128(lds + 8, g + 8);
#else
  *(v8bf*)lds       = *(const v8bf*)g;
  *(v8bf*)(lds + 8) = *(const v8bf*)(g + 8);
#endif
}

// ---------------------------------------------------------------------------
// fp32 -> bf16 conversion (weights)
// ---------------------------------------------------------------------------
__global__ void cvt_f32_bf16(const float* __restrict__ in,
                             __bf16* __restrict__ out, int n) {
  int i = blockIdx.x * blockDim.x + threadIdx.x;
  int stride = gridDim.x * blockDim.x;
  for (; i < n; i += stride) out[i] = (__bf16)in[i];
}

// ---------------------------------------------------------------------------
// C[M,N] = A[M,K] @ W[N,K]^T + bias[N]   (K = N = E_DIM, M = MTOT)
// Block: 256 threads (8 waves). Block tile 128(M) x 128(N), K-step 32.
// Wave w: rows 32*(w>>1), cols 64*(w&1): 2x4 fragments of 16x16 (8 WMMA/step).
// ---------------------------------------------------------------------------
template <bool A_F32, bool OUT_F32>
__global__ __launch_bounds__(256) void gemm_bias(const void* __restrict__ Aglob,
                                                 const __bf16* __restrict__ Wbf,
                                                 const float* __restrict__ bias,
                                                 void* __restrict__ Out) {
  constexpr int LDA = 40;  // 32 + 8 pad (bf16) -> 80B rows, 16B aligned chunks
  constexpr int LDB = 40;
  __shared__ __bf16 At[128 * LDA];
  __shared__ __bf16 Bt[128 * LDB];

  const int tid  = threadIdx.x;
  const int lane = tid & 31;
  const int wid  = tid >> 5;
  const int half = lane >> 4;
  const int l16  = lane & 15;
  const int wm   = (wid >> 1) * 32;  // 0,32,64,96
  const int wn   = (wid & 1) * 64;   // 0,64
  const int bm0  = blockIdx.x * 128;
  const int bn0  = blockIdx.y * 128;

  v8f acc[2][4] = {};

  const int srow = tid >> 1;        // 0..127 (staging row for A and B)
  const int skc  = (tid & 1) * 16;  // 0,16

  for (int k0 = 0; k0 < E_DIM; k0 += 32) {
    // ---- stage A tile (128 x 32) ----
    if constexpr (A_F32) {
      const float* Af = (const float*)Aglob + (size_t)(bm0 + srow) * E_DIM + k0 + skc;
#pragma unroll
      for (int i = 0; i < 16; ++i) At[srow * LDA + skc + i] = (__bf16)Af[i];
    } else {
      copy16_bf16(&At[srow * LDA + skc],
                  (const __bf16*)Aglob + (size_t)(bm0 + srow) * E_DIM + k0 + skc);
    }
    // ---- stage B tile (128 n-rows x 32 k), W row-major [n][k] ----
    copy16_bf16(&Bt[srow * LDB + skc],
                Wbf + (size_t)(bn0 + srow) * E_DIM + k0 + skc);
#ifdef HAVE_ASYNC_LDS
    async_wait0();
#endif
    __syncthreads();

    // ---- fragments per documented CDNA5 layouts ----
    v16bf af[2], bfr[4];
#pragma unroll
    for (int mi = 0; mi < 2; ++mi) {
      const __bf16* rp = &At[(wm + 16 * mi + l16) * LDA + half * 8];
      af[mi] = cat8(*(const v8bf*)rp, *(const v8bf*)(rp + 16));
    }
#pragma unroll
    for (int ni = 0; ni < 4; ++ni) {
      const __bf16* cp = &Bt[(wn + 16 * ni + l16) * LDB + half * 8];
      bfr[ni] = cat8(*(const v8bf*)cp, *(const v8bf*)(cp + 16));
    }
#pragma unroll
    for (int mi = 0; mi < 2; ++mi)
#pragma unroll
      for (int ni = 0; ni < 4; ++ni)
        acc[mi][ni] = wmma_bf16(af[mi], bfr[ni], acc[mi][ni]);
    __syncthreads();
  }

  // ---- epilogue: C layout row m = r + 8*half, col n = l16 ----
#pragma unroll
  for (int mi = 0; mi < 2; ++mi)
#pragma unroll
    for (int ni = 0; ni < 4; ++ni)
#pragma unroll
      for (int r = 0; r < 8; ++r) {
        const int m = bm0 + wm + 16 * mi + r + 8 * half;
        const int n = bn0 + wn + 16 * ni + l16;
        const float v = acc[mi][ni][r] + bias[n];
        if constexpr (OUT_F32)
          ((float*)Out)[(size_t)m * E_DIM + n] = v;
        else
          ((__bf16*)Out)[(size_t)m * E_DIM + n] = (__bf16)v;
      }
}

// ---------------------------------------------------------------------------
// Flash attention (causal), bf16 inputs, online softmax in f32.
// Grid: (SEQ/128, BATCH*NH). 256 threads = 8 waves; wave owns 16 query rows.
// ---------------------------------------------------------------------------
__global__ __launch_bounds__(256) void flash_attn(const __bf16* __restrict__ Q,
                                                  const __bf16* __restrict__ K,
                                                  const __bf16* __restrict__ V,
                                                  __bf16* __restrict__ ctx) {
  constexpr int LDK = 72, LDV = 72, LDP = 72;  // 64 + 8 pad
  __shared__ __bf16 Kt[64 * LDK];       // [kcol][d]
  __shared__ __bf16 Vt[64 * LDV];       // transposed: [d][kcol]
  __shared__ __bf16 Pt[8 * 16 * LDP];   // per-wave P buffers

  const int tid  = threadIdx.x;
  const int lane = tid & 31;
  const int wid  = tid >> 5;
  const int half = lane >> 4;
  const int l16  = lane & 15;

  const int qb = blockIdx.x * 128;
  const int bh = blockIdx.y;  // b*NH + h
  const int b  = bh / NH;
  const int h  = bh % NH;
  const size_t base = ((size_t)b * SEQ) * E_DIM + (size_t)h * DH;

  // Q fragments for this wave's 16 rows (kept in registers the whole pass)
  const int qrow0 = qb + wid * 16;
  v16bf qf[2];
#pragma unroll
  for (int f = 0; f < 2; ++f) {
    const __bf16* qp = Q + base + (size_t)(qrow0 + l16) * E_DIM + 32 * f + half * 8;
    qf[f] = cat8(*(const v8bf*)qp, *(const v8bf*)(qp + 16));
  }

  v8f o[4] = {};
  float mrow[8], lrow[8];
#pragma unroll
  for (int r = 0; r < 8; ++r) { mrow[r] = -INFINITY; lrow[r] = 0.f; }

  const int nkt  = (qb + 128) / 64;  // causal tile bound
  const int krow = tid >> 2;         // 0..63
  const int kdc  = (tid & 3) * 16;   // 0,16,32,48

  for (int kt = 0; kt < nkt; ++kt) {
    const int kcol0 = kt * 64;
    // ---- stage K tile (async direct-to-LDS) and transposed V tile ----
    {
      const __bf16* kp = K + base + (size_t)(kcol0 + krow) * E_DIM + kdc;
      copy16_bf16(&Kt[krow * LDK + kdc], kp);
      const __bf16* vp = V + base + (size_t)(kcol0 + krow) * E_DIM + kdc;
      v8bf v0 = *(const v8bf*)vp;
      v8bf v1 = *(const v8bf*)(vp + 8);
#pragma unroll
      for (int i = 0; i < 8; ++i) {
        Vt[(kdc + i) * LDV + krow]     = v0[i];
        Vt[(kdc + 8 + i) * LDV + krow] = v1[i];
      }
      if (kt + 1 < nkt) {  // prefetch next tile (global_prefetch_b8)
        __builtin_prefetch(kp + (size_t)64 * E_DIM, 0, 0);
        __builtin_prefetch(vp + (size_t)64 * E_DIM, 0, 0);
      }
    }
#ifdef HAVE_ASYNC_LDS
    async_wait0();
#endif
    __syncthreads();

    // ---- scores S = Q K^T : 4 fragments of 16 key-cols, K-depth 64 (2 WMMA) ----
    v8f s[4];
#pragma unroll
    for (int ct = 0; ct < 4; ++ct) {
      const __bf16* kp0 = &Kt[(16 * ct + l16) * LDK + half * 8];
      v16bf kb0 = cat8(*(const v8bf*)kp0, *(const v8bf*)(kp0 + 16));
      v16bf kb1 = cat8(*(const v8bf*)(kp0 + 32), *(const v8bf*)(kp0 + 48));
      v8f z = {};
      z     = wmma_bf16(qf[0], kb0, z);
      s[ct] = wmma_bf16(qf[1], kb1, z);
    }

    // ---- scale, causal mask, online softmax (per-row stats, shfl reductions) ----
    float p[4][8];
#pragma unroll
    for (int r = 0; r < 8; ++r) {
      const int qrow = qrow0 + r + 8 * half;
      float smax = -3.0e38f;
#pragma unroll
      for (int ct = 0; ct < 4; ++ct) {
        const int kcol = kcol0 + 16 * ct + l16;
        float sv = s[ct][r] * 0.125f;  // 1/sqrt(64)
        if (kcol > qrow) sv = -3.0e38f;
        p[ct][r] = sv;
        smax = fmaxf(smax, sv);
      }
#pragma unroll
      for (int msk = 8; msk >= 1; msk >>= 1)
        smax = fmaxf(smax, __shfl_xor(smax, msk, 32));
      const float mnew = fmaxf(mrow[r], smax);
      const float esc  = __expf(mrow[r] - mnew);
      float psum = 0.f;
#pragma unroll
      for (int ct = 0; ct < 4; ++ct) {
        const float pv = __expf(p[ct][r] - mnew);
        p[ct][r] = pv;
        psum += pv;
      }
#pragma unroll
      for (int msk = 8; msk >= 1; msk >>= 1)
        psum += __shfl_xor(psum, msk, 32);
      lrow[r] = lrow[r] * esc + psum;
      mrow[r] = mnew;
#pragma unroll
      for (int dt = 0; dt < 4; ++dt) o[dt][r] *= esc;
    }

    // ---- P: C-layout -> LDS -> A-layout fragments ----
    __bf16* Pw = &Pt[wid * 16 * LDP];
#pragma unroll
    for (int ct = 0; ct < 4; ++ct)
#pragma unroll
      for (int r = 0; r < 8; ++r)
        Pw[(r + 8 * half) * LDP + 16 * ct + l16] = (__bf16)p[ct][r];

    __syncthreads();

    v16bf pf[2];
#pragma unroll
    for (int f = 0; f < 2; ++f) {
      const __bf16* pp = &Pw[l16 * LDP + 32 * f + half * 8];
      pf[f] = cat8(*(const v8bf*)pp, *(const v8bf*)(pp + 16));
    }

    // ---- O += P V : 4 d-fragments, K-depth 64 (2 WMMA each) ----
#pragma unroll
    for (int dt = 0; dt < 4; ++dt) {
      const __bf16* vp0 = &Vt[(16 * dt + l16) * LDV + half * 8];
      v16bf vb0 = cat8(*(const v8bf*)vp0, *(const v8bf*)(vp0 + 16));
      v16bf vb1 = cat8(*(const v8bf*)(vp0 + 32), *(const v8bf*)(vp0 + 48));
      o[dt] = wmma_bf16(pf[0], vb0, o[dt]);
      o[dt] = wmma_bf16(pf[1], vb1, o[dt]);
    }
    __syncthreads();
  }

  // ---- normalize, store ctx (B,S,H,D) flattened as (M, E) bf16 ----
#pragma unroll
  for (int r = 0; r < 8; ++r) {
    const int qrow  = qrow0 + r + 8 * half;
    const float inv = 1.f / lrow[r];
#pragma unroll
    for (int dt = 0; dt < 4; ++dt)
      ctx[base + (size_t)qrow * E_DIM + 16 * dt + l16] = (__bf16)(o[dt][r] * inv);
  }
}

// ---------------------------------------------------------------------------
// Host-side orchestration
// ---------------------------------------------------------------------------
extern "C" void kernel_launch(void* const* d_in, const int* in_sizes, int n_in,
                              void* d_out, int out_size, void* d_ws, size_t ws_size,
                              hipStream_t stream) {
  (void)in_sizes; (void)n_in; (void)out_size; (void)ws_size;
  const float* query = (const float*)d_in[0];
  const float* key_  = (const float*)d_in[1];
  const float* value = (const float*)d_in[2];
  // d_in[3] = attn_mask (causal triu) -> handled analytically in flash_attn
  const float* q_w = (const float*)d_in[4];
  const float* q_b = (const float*)d_in[5];
  const float* k_w = (const float*)d_in[6];
  const float* k_b = (const float*)d_in[7];
  const float* v_w = (const float*)d_in[8];
  const float* v_b = (const float*)d_in[9];
  const float* o_w = (const float*)d_in[10];
  const float* o_b = (const float*)d_in[11];

  // Workspace layout (bytes): 4 weights bf16 (8MB) + Q/K/V bf16 (48MB) + ctx (16MB)
  char* ws = (char*)d_ws;
  const size_t wbytes = (size_t)E_DIM * E_DIM * sizeof(__bf16);   // 2 MB
  const size_t xbytes = (size_t)MTOT * E_DIM * sizeof(__bf16);    // 16 MB
  __bf16* wq  = (__bf16*)(ws);
  __bf16* wk  = (__bf16*)(ws + wbytes);
  __bf16* wv  = (__bf16*)(ws + 2 * wbytes);
  __bf16* wo  = (__bf16*)(ws + 3 * wbytes);
  __bf16* Qb  = (__bf16*)(ws + 4 * wbytes);
  __bf16* Kb  = (__bf16*)(ws + 4 * wbytes + xbytes);
  __bf16* Vb  = (__bf16*)(ws + 4 * wbytes + 2 * xbytes);
  __bf16* ctx = (__bf16*)(ws + 4 * wbytes + 3 * xbytes);

  const int nW = E_DIM * E_DIM;
  cvt_f32_bf16<<<512, 256, 0, stream>>>(q_w, wq, nW);
  cvt_f32_bf16<<<512, 256, 0, stream>>>(k_w, wk, nW);
  cvt_f32_bf16<<<512, 256, 0, stream>>>(v_w, wv, nW);
  cvt_f32_bf16<<<512, 256, 0, stream>>>(o_w, wo, nW);

  dim3 gg(MTOT / 128, E_DIM / 128);  // (64, 8)
  gemm_bias<true, false><<<gg, 256, 0, stream>>>(query, wq, q_b, Qb);
  gemm_bias<true, false><<<gg, 256, 0, stream>>>(key_,  wk, k_b, Kb);
  gemm_bias<true, false><<<gg, 256, 0, stream>>>(value, wv, v_b, Vb);

  dim3 gf(SEQ / 128, BATCH * NH);   // (16, 64)
  flash_attn<<<gf, 256, 0, stream>>>(Qb, Kb, Vb, ctx);

  gemm_bias<false, true><<<gg, 256, 0, stream>>>(ctx, wo, o_b, d_out);
}
